// SelfMatchingAttention_58488864637420
// MI455X (gfx1250) — matrix operand order repaired and checked by reading, not compiled, not run on hardware
//
#include <hip/hip_runtime.h>
#include <hip/hip_bf16.h>

// ---------------------------------------------------------------------------
// SelfMatchingAttention + BiGRU for MI455X (gfx1250, wave32, WMMA)
//   L=1024, B=64, H=128, 2H=256, 3H=384, 4H=512
//
// Pipeline (reverse attention == flipped forward attention, so attention is
// computed once):
//   K1a: EW   = exp(2*clamp(passEnc @ Wp_w.T  + Wp_b))   stored [b][h][l]   (WMMA f32)
//   K1b: EX   = exp(2*clamp(passEnc @ Wph_w.T + Wph_b))  stored [t][b][h]   (WMMA f32)
//   K2 : scores/softmax/context fused, tanh via 1-2*rcp(EW*EX+1)   -> CC [t][b][512]
//   K3 : CCG  = sigmoid(CC @ Wg_w.T + Wg_b) * CC                           (WMMA f32)
//   K4 : GIF  = CCG       @ wih_f.T + bih_f                                (WMMA f32)
//   K5 : GIR  = flip(CCG) @ wih_r.T + bih_r                                (WMMA f32)
//   K6 : persistent sequential GRU, gh = h @ whh.T via WMMA bf16 16x16x32
// ---------------------------------------------------------------------------

typedef __attribute__((ext_vector_type(2)))  float  v2f;
typedef __attribute__((ext_vector_type(8)))  float  v8f;
typedef __attribute__((ext_vector_type(16))) __bf16 v16bf;

static constexpr int SEQL = 1024;
static constexpr int NB   = 64;

__device__ __forceinline__ float fast_sig(float x) {
    return __builtin_amdgcn_rcpf(1.0f + __expf(-x));
}
__device__ __forceinline__ float fast_tanh(float x) {
    float c  = fminf(fmaxf(x, -15.0f), 15.0f);
    float e2 = __expf(2.0f * c);
    return 1.0f - 2.0f * __builtin_amdgcn_rcpf(e2 + 1.0f);
}

// ---------------------------------------------------------------------------
// Generic f32 WMMA GEMM:  D = epilogue(A(MxK) @ Bm(NxK)^T + bias)
// block = 256 threads (8 waves). Each wave owns one 16x16 tile:
//   m0 = blockIdx.x*16,  n0 = (blockIdx.y*8 + wave)*16
// modes: 0 plain store  | 1 exp(2*clamp()) transposed [b][h][l] store
//        2 gated: sigmoid()*Cmul | 3 plain store with A rows flipped over l
//        4 exp(2*clamp()) plain store
// ---------------------------------------------------------------------------
__global__ __launch_bounds__(256)
void gemm_wmma_f32(const float* __restrict__ A, const float* __restrict__ Bm,
                   const float* __restrict__ bias, float* __restrict__ D,
                   const float* __restrict__ Cmul, int K, int ldd, int mode)
{
    const int lane = threadIdx.x & 31;
    const int wave = threadIdx.x >> 5;
    const int r    = lane & 15;        // row-in-tile
    const int grp  = lane >> 4;        // K half-group
    const int m0   = blockIdx.x * 16;
    const int n0   = (blockIdx.y * 8 + wave) * 16;

    int arow = m0 + r;
    if (mode == 3) {                   // flip over sequence dim (rows are l*64+b)
        int l = arow >> 6, bb = arow & 63;
        arow = (SEQL - 1 - l) * 64 + bb;
    }
    const float* ap = A  + (size_t)arow      * K + grp * 2;
    const float* bp = Bm + (size_t)(n0 + r)  * K + grp * 2;

    v8f acc = {};
    for (int k = 0; k < K; k += 4) {
        v2f av = *(const v2f*)(ap + k);    // lanes 0-15: K0,K1 ; lanes 16-31: K2,K3
        v2f bv = *(const v2f*)(bp + k);
        acc = __builtin_amdgcn_wmma_f32_16x16x4_f32(
                  false, av, false, bv, (short)0, acc, false, false);
    }

    const int   n  = n0 + r;
    const float bn = bias[n];
#pragma unroll
    for (int i = 0; i < 8; ++i) {
        const int m = m0 + i + grp * 8;          // C layout: lane<16 -> M=i, else M=i+8
        float v = acc[i] + bn;
        if (mode == 0 || mode == 3) {
            D[(size_t)m * ldd + n] = v;
        } else if (mode == 1) {                  // EW: [b][h][l], pre-exponentiated
            float c = fminf(fmaxf(v, -15.0f), 15.0f);
            int l = m >> 6, bb = m & 63;
            D[((size_t)(bb * 128 + n)) * SEQL + l] = __expf(2.0f * c);
        } else if (mode == 4) {                  // EX: [t][b][h], pre-exponentiated
            float c = fminf(fmaxf(v, -15.0f), 15.0f);
            D[(size_t)m * ldd + n] = __expf(2.0f * c);
        } else {                                 // mode 2: gated in separate buffer
            float g = fast_sig(v);
            D[(size_t)m * ldd + n] = g * Cmul[(size_t)m * ldd + n];
        }
    }
}

// ---------------------------------------------------------------------------
// K2: fused scores + softmax + context.  One block per (8 t's, one b).
//   scores[t,l] = sum_h (1 - 2/(EW[b,h,l]*EX[t,b,h] + 1)) * v[b,h]
//   a = softmax_l(scores);  c[t,d] = sum_l a[t,l] * passEnc[l,b,d]
//   CC[t,b,0:256] = passEnc[t,b,:],  CC[t,b,256:512] = c
// ---------------------------------------------------------------------------
__global__ __launch_bounds__(256)
void attn_score_ctx(const float* __restrict__ EW,   // [B][H][L]
                    const float* __restrict__ EX,   // [T][B][H]
                    const float* __restrict__ Vt,   // [B][H]
                    const float* __restrict__ pe,   // [L][B][256]
                    float* __restrict__ CC)         // [T][B][512]
{
    __shared__ float xp[8][128];
    __shared__ float vv[128];
    __shared__ float sc[8][SEQL];

    const int b   = blockIdx.y;
    const int t0  = blockIdx.x * 8;
    const int tid = threadIdx.x;

    for (int i = tid; i < 8 * 128; i += 256)
        xp[i >> 7][i & 127] = EX[((size_t)(t0 + (i >> 7)) * NB + b) * 128 + (i & 127)];
    if (tid < 128) vv[tid] = Vt[b * 128 + tid];
    __syncthreads();

    // ---- pass 1: scores (the 8.6G-op hot loop: mul, add, rcp, fma, fma) ----
    for (int j = 0; j < 4; ++j) {
        const int l = tid + 256 * j;                     // coalesced over lanes
        float acc[8] = {};
        const float* ewp = EW + (size_t)b * 128 * SEQL + l;
        for (int h = 0; h < 128; ++h) {
            const float ew = ewp[(size_t)h * SEQL];
            const float vh = vv[h];
#pragma unroll
            for (int t = 0; t < 8; ++t) {
                float m  = ew * xp[t][h];
                float th = 1.0f - 2.0f * __builtin_amdgcn_rcpf(m + 1.0f);
                acc[t] = fmaf(th, vh, acc[t]);
            }
        }
#pragma unroll
        for (int t = 0; t < 8; ++t) sc[t][l] = acc[t];
    }
    __syncthreads();

    // ---- softmax: wave w owns row t=w (wave32 shuffle reductions) ----
    {
        const int t = tid >> 5, lane = tid & 31;
        float mx = -1e30f;
        for (int l = lane; l < SEQL; l += 32) mx = fmaxf(mx, sc[t][l]);
        for (int off = 16; off; off >>= 1) mx = fmaxf(mx, __shfl_xor(mx, off, 32));
        float sum = 0.0f;
        for (int l = lane; l < SEQL; l += 32) {
            float e = __expf(sc[t][l] - mx);
            sc[t][l] = e;
            sum += e;
        }
        for (int off = 16; off; off >>= 1) sum += __shfl_xor(sum, off, 32);
        const float inv = __builtin_amdgcn_rcpf(sum);
        for (int l = lane; l < SEQL; l += 32) sc[t][l] *= inv;
    }
    __syncthreads();

    // ---- pass 2: context accumulation, d = tid (coalesced), a broadcast ----
    const int d = tid;
    float cacc[8] = {};
    for (int l = 0; l < SEQL; ++l) {
        const float p = pe[((size_t)l * NB + b) * 256 + d];
#pragma unroll
        for (int t = 0; t < 8; ++t) cacc[t] = fmaf(sc[t][l], p, cacc[t]);
    }
#pragma unroll
    for (int t = 0; t < 8; ++t) {
        const size_t row = (size_t)(t0 + t) * NB + b;
        CC[row * 512 + 256 + d] = cacc[t];
        CC[row * 512 + d]       = pe[row * 256 + d];    // concat x_t
    }
}

// ---------------------------------------------------------------------------
// K6: persistent sequential GRU. One block per direction; h lives in LDS.
//   gh = h(64x128) @ whh^T(384x128) via V_WMMA_F32_16X16X32_BF16
//   8 waves x (3 n-tiles x 4 m-tiles) = 96 tiles, 4 k-steps each.
// ---------------------------------------------------------------------------
__global__ __launch_bounds__(256)
void gru_seq(const float* __restrict__ GIF, const float* __restrict__ GIR,
             const float* __restrict__ whh_f, const float* __restrict__ whh_r,
             const float* __restrict__ bhh_f, const float* __restrict__ bhh_r,
             const float* __restrict__ h0f,   const float* __restrict__ h0r,
             __bf16* __restrict__ whhbf_ws,   float* __restrict__ ghbuf_ws,
             float* __restrict__ out)
{
    __shared__ float  hsh[64 * 128];
    __shared__ __bf16 hbf[64 * 128];
    __shared__ float  bsh[384];

    const int dir = blockIdx.x;
    const float* GI  = dir ? GIR   : GIF;
    const float* whh = dir ? whh_r : whh_f;
    const float* bhh = dir ? bhh_r : bhh_f;
    const float* h0  = dir ? h0r   : h0f;
    __bf16* whhbf = whhbf_ws + (size_t)dir * 384 * 128;
    float*  gh    = ghbuf_ws + (size_t)dir * 64 * 384;
    const int outcol = dir * 128;

    const int tid  = threadIdx.x;
    const int lane = tid & 31, wave = tid >> 5;
    const int r    = lane & 15, grp = lane >> 4;

    for (int i = tid; i < 384 * 128; i += 256) whhbf[i] = (__bf16)whh[i];
    for (int i = tid; i < 64 * 128;  i += 256) {
        float h = h0[i]; hsh[i] = h; hbf[i] = (__bf16)h;
    }
    for (int i = tid; i < 384; i += 256) bsh[i] = bhh[i];
    __threadfence_block();
    __syncthreads();

    for (int t = 0; t < SEQL; ++t) {
        // ---- gh = h @ whh^T (bf16 WMMA) ----
#pragma unroll
        for (int nt3 = 0; nt3 < 3; ++nt3) {
            const int nt = wave + nt3 * 8;
            const int n  = nt * 16 + r;
#pragma unroll
            for (int mt = 0; mt < 4; ++mt) {
                v8f acc = {};
#pragma unroll
                for (int kk = 0; kk < 4; ++kk) {
                    const int k0 = kk * 32;
                    v16bf av, bv;
                    const int arow = mt * 16 + r;
#pragma unroll
                    for (int i = 0; i < 16; ++i) {        // A 16x32 bf16 striping
                        int ka = k0 + grp * 16 + i;
                        av[i] = hbf[arow * 128 + ka];
                    }
                    const __bf16* bp = whhbf + (size_t)n * 128 + k0 + grp * 16;
#pragma unroll
                    for (int i = 0; i < 16; ++i) bv[i] = bp[i];  // B 32x16 striping
                    acc = __builtin_amdgcn_wmma_f32_16x16x32_bf16(
                              false, av, false, bv, (short)0, acc, false, false);
                }
#pragma unroll
                for (int i = 0; i < 8; ++i) {
                    const int m = mt * 16 + i + grp * 8;
                    gh[m * 384 + n] = acc[i];
                }
            }
        }
        __threadfence_block();
        __syncthreads();

        // ---- gates + state update ----
        const size_t rowbase = (size_t)t * NB * 384;
        for (int idx = tid; idx < 64 * 128; idx += 256) {
            const int bb = idx >> 7, k = idx & 127;
            const size_t gir = rowbase + (size_t)bb * 384;
            const float ir  = GI[gir + k];
            const float iz  = GI[gir + 128 + k];
            const float inn = GI[gir + 256 + k];
            const float hr  = gh[bb * 384 + k]       + bsh[k];
            const float hz  = gh[bb * 384 + 128 + k] + bsh[128 + k];
            const float hn  = gh[bb * 384 + 256 + k] + bsh[256 + k];
            const float rg  = fast_sig(ir + hr);
            const float zg  = fast_sig(iz + hz);
            const float ng  = fast_tanh(inn + rg * hn);
            const float hprev = hsh[idx];
            const float hnew  = (1.0f - zg) * ng + zg * hprev;
            hsh[idx] = hnew;
            hbf[idx] = (__bf16)hnew;
            out[((size_t)t * NB + bb) * 256 + outcol + k] = hnew;
        }
        __threadfence_block();
        __syncthreads();
    }
}

// ---------------------------------------------------------------------------
extern "C" void kernel_launch(void* const* d_in, const int* in_sizes, int n_in,
                              void* d_out, int out_size, void* d_ws, size_t ws_size,
                              hipStream_t stream)
{
    (void)in_sizes; (void)n_in; (void)out_size; (void)ws_size;

    const float* passEnc = (const float*)d_in[0];
    const float* Wp_w    = (const float*)d_in[1];
    const float* Wp_b    = (const float*)d_in[2];
    const float* Wph_w   = (const float*)d_in[3];
    const float* Wph_b   = (const float*)d_in[4];
    const float* Wg_w    = (const float*)d_in[5];
    const float* Wg_b    = (const float*)d_in[6];
    const float* Vt      = (const float*)d_in[7];
    const float* wih_f   = (const float*)d_in[8];
    const float* whh_f   = (const float*)d_in[9];
    const float* bih_f   = (const float*)d_in[10];
    const float* bhh_f   = (const float*)d_in[11];
    const float* wih_r   = (const float*)d_in[12];
    const float* whh_r   = (const float*)d_in[13];
    const float* bih_r   = (const float*)d_in[14];
    const float* bhh_r   = (const float*)d_in[15];
    const float* h0f     = (const float*)d_in[16];
    const float* h0r     = (const float*)d_in[17];
    float* out = (float*)d_out;

    // workspace partition (floats)
    float* w   = (float*)d_ws;
    float* EW  = w;                                      // 64*128*1024
    float* EX  = EW  + (size_t)64 * 128 * 1024;          // 1024*64*128
    float* CCb = EX  + (size_t)1024 * 64 * 128;          // 1024*64*512 (raw cc)
    float* CCG = CCb + (size_t)1024 * 64 * 512;          // 1024*64*512 (gated cc)
    float* GIF = CCG + (size_t)1024 * 64 * 512;          // 1024*64*384
    float* GIR = GIF + (size_t)1024 * 64 * 384;          // 1024*64*384
    float* GH  = GIR + (size_t)1024 * 64 * 384;          // 2*64*384
    __bf16* WHHBF = (__bf16*)(GH + 2 * 64 * 384);        // 2*384*128 bf16

    const dim3 blk(256);
    // K1a: EW = exp(2*clamp(passEnc @ Wp_w^T + Wp_b)) transposed [b][h][l]
    gemm_wmma_f32<<<dim3(4096, 1), blk, 0, stream>>>(passEnc, Wp_w,  Wp_b,  EW,  nullptr, 256, 0,   1);
    // K1b: EX = exp(2*clamp(passEnc @ Wph_w^T + Wph_b)) [t][b][h]
    gemm_wmma_f32<<<dim3(4096, 1), blk, 0, stream>>>(passEnc, Wph_w, Wph_b, EX,  nullptr, 256, 128, 4);
    // K2: fused attention (scores -> softmax -> context -> concat)
    attn_score_ctx<<<dim3(128, 64), blk, 0, stream>>>(EW, EX, Vt, passEnc, CCb);
    // K3: gated cc = sigmoid(cc @ Wg^T + b) * cc
    gemm_wmma_f32<<<dim3(4096, 4), blk, 0, stream>>>(CCb, Wg_w, Wg_b, CCG, CCb, 512, 512, 2);
    // K4/K5: input-side GRU GEMMs for all timesteps (reverse uses flipped rows)
    gemm_wmma_f32<<<dim3(4096, 3), blk, 0, stream>>>(CCG, wih_f, bih_f, GIF, nullptr, 512, 384, 0);
    gemm_wmma_f32<<<dim3(4096, 3), blk, 0, stream>>>(CCG, wih_r, bih_r, GIR, nullptr, 512, 384, 3);
    // K6: sequential GRU (persistent, one block per direction)
    gru_seq<<<dim3(2), blk, 0, stream>>>(GIF, GIR, whh_f, whh_r, bhh_f, bhh_r,
                                         h0f, h0r, WHHBF, GH, out);
}
